// BasicBlock_61658550501925
// MI455X (gfx1250) — matrix operand order repaired and checked by reading, not compiled, not run on hardware
//
#include <hip/hip_runtime.h>
#include <hip/hip_bf16.h>
#include <cmath>

typedef __attribute__((ext_vector_type(16))) _Float16 v16h;
typedef __attribute__((ext_vector_type(8)))  float    v8f;

#define WMMA_F16(a, b, c) \
    __builtin_amdgcn_wmma_f32_16x16x32_f16(false, (a), false, (b), (short)0, (c), false, false)

__device__ __forceinline__ float gelu_f(float x) {
    return 0.5f * x * (1.0f + erff(x * 0.70710678118654752440f));
}

// ---------------- WMMA fragment loaders (CDNA5 16x16x32 f16 layouts) ----------------
// A (16x32 f16): lane L -> row m0+(L&15); half=L>>4; slot s = K (16*half + s).
__device__ __forceinline__ v16h frag_a(const _Float16* base, int m0, int k0, int ld) {
    int lane = threadIdx.x & 31;
    const _Float16* p = base + (m0 + (lane & 15)) * ld + k0 + ((lane >> 4) << 4);
    v16h a;
#pragma unroll
    for (int s = 0; s < 16; ++s) a[s] = p[s];
    return a;
}

// A built from fp32 LDS where only K<16 is valid (K-dim 16 zero-padded to 32).
__device__ __forceinline__ v16h frag_a_f32_k16(const float* base, int m0, int k0, int ld) {
    int lane = threadIdx.x & 31;
    int hf = lane >> 4;
    const float* p = base + (m0 + (lane & 15)) * ld + k0;
    v16h a;
#pragma unroll
    for (int s = 0; s < 16; ++s) a[s] = hf ? (_Float16)0.0f : (_Float16)p[s];   // K = 16*half + s; K >= 16 -> zero pad
    return a;
}

// B (32x16 f16) from row-major W[n][k] (ld elems per row): lane L -> col n0+(L&15);
// slot s -> K = 16*(L>>4) + s, element W[n][k0+K] (contiguous in k).
__device__ __forceinline__ v16h frag_b(const _Float16* base, int n0, int k0, int ld) {
    int lane = threadIdx.x & 31;
    const _Float16* p = base + (n0 + (lane & 15)) * ld + k0 + ((lane >> 4) << 4);
    v16h b;
#pragma unroll
    for (int s = 0; s < 16; ++s) b[s] = p[s];
    return b;
}

// B where only K<16 is valid (zero-padded to 32): lanes 16..31 (K=16..31) are zero.
__device__ __forceinline__ v16h frag_b_k16(const _Float16* base, int n0, int k0, int ld) {
    int lane = threadIdx.x & 31;
    int hf = lane >> 4;
    const _Float16* p = base + (n0 + (lane & 15)) * ld + k0;
    v16h b;
#pragma unroll
    for (int s = 0; s < 16; ++s) b[s] = hf ? (_Float16)0.0f : p[s];
    return b;
}

// ---------------- K1/K3: per-plane LayerNorm stats (mean, rstd over H*W) ----------------
__global__ void k_plane_stats(const float* __restrict__ src, float* __restrict__ mu,
                              float* __restrict__ rs, int plane_elems) {
    __shared__ float s1[256];
    __shared__ float s2[256];
    int p = blockIdx.x;
    const float* base = src + (size_t)p * plane_elems;
    float s = 0.f, ss = 0.f;
    for (int i = threadIdx.x; i < plane_elems; i += blockDim.x) {
        float v = base[i];
        s += v; ss += v * v;
    }
    s1[threadIdx.x] = s; s2[threadIdx.x] = ss;
    __syncthreads();
    for (int off = 128; off > 0; off >>= 1) {
        if ((int)threadIdx.x < off) {
            s1[threadIdx.x] += s1[threadIdx.x + off];
            s2[threadIdx.x] += s2[threadIdx.x + off];
        }
        __syncthreads();
    }
    if (threadIdx.x == 0) {
        float inv = 1.0f / (float)plane_elems;
        float m = s1[0] * inv;
        float var = s2[0] * inv - m * m;
        mu[p] = m;
        rs[p] = rsqrtf(var + 1e-5f);
    }
}

// ---------------- K2: fused LN1-apply + dual 1x1 conv (two WMMA GEMMs) ----------------
// 4096 workgroups x 128 threads; each WG handles 64 tokens (fixed b,h; w in [w0,w0+64)).
__global__ void k_lclg_conv(const float* __restrict__ x,
                            const float* __restrict__ g1a, const float* __restrict__ b1a,
                            const float* __restrict__ cw1, const float* __restrict__ cb1,
                            const float* __restrict__ g1b, const float* __restrict__ b1b,
                            const float* __restrict__ cw2, const float* __restrict__ cb2,
                            const float* __restrict__ mu1, const float* __restrict__ rs1,
                            float* __restrict__ y1raw, float* __restrict__ y2raw) {
    __shared__ _Float16 zh1[64 * 64];
    __shared__ _Float16 zh2[64 * 64];
    __shared__ _Float16 wbuf[64 * 64];
    __shared__ float outs[64 * 64];

    int wg = blockIdx.x;
    int b = wg >> 8;
    int r = wg & 255;
    int h = r >> 1;
    int w0 = (r & 1) << 6;
    int tid = threadIdx.x;
    int lane = tid & 31;
    int wv = tid >> 5;

    // Build z1, z2 (token-major [n][c]) with LN1 affine (stats shared: same x).
    for (int idx = tid; idx < 4096; idx += 128) {
        int c = idx >> 6, n = idx & 63;
        int hw = h * 128 + w0 + n;
        int pc = b * 64 + c;
        float xn = (x[(size_t)pc * 16384 + hw] - mu1[pc]) * rs1[pc];
        zh1[n * 64 + c] = (_Float16)(xn * g1a[hw] + b1a[hw]);
        zh2[n * 64 + c] = (_Float16)(xn * g1b[hw] + b1b[hw]);
    }
    for (int idx = tid; idx < 4096; idx += 128) wbuf[idx] = (_Float16)cw1[idx];
    __syncthreads();

    int m0 = wv * 16;
    int Nn = lane & 15, mh = lane >> 4;
    v8f vz = {};

    // Branch 1 GEMM
    {
        v8f acc[4];
#pragma unroll
        for (int nt = 0; nt < 4; ++nt) acc[nt] = vz;
#pragma unroll
        for (int kc = 0; kc < 2; ++kc) {
            v16h a = frag_a(zh1, m0, kc * 32, 64);
#pragma unroll
            for (int nt = 0; nt < 4; ++nt) {
                v16h bf = frag_b(wbuf, nt * 16, kc * 32, 64);
                acc[nt] = WMMA_F16(a, bf, acc[nt]);
            }
        }
#pragma unroll
        for (int nt = 0; nt < 4; ++nt)
#pragma unroll
            for (int r8 = 0; r8 < 8; ++r8) {
                int M = m0 + r8 + 8 * mh, O = nt * 16 + Nn;
                outs[M * 64 + O] = acc[nt][r8] + cb1[O];
            }
    }
    __syncthreads();
    for (int idx = tid; idx < 4096; idx += 128) {
        int c = idx >> 6, n = idx & 63;
        y1raw[(size_t)(b * 64 + c) * 16384 + h * 128 + w0 + n] = outs[n * 64 + c];
    }
    for (int idx = tid; idx < 4096; idx += 128) wbuf[idx] = (_Float16)cw2[idx];
    __syncthreads();

    // Branch 2 GEMM
    {
        v8f acc[4];
#pragma unroll
        for (int nt = 0; nt < 4; ++nt) acc[nt] = vz;
#pragma unroll
        for (int kc = 0; kc < 2; ++kc) {
            v16h a = frag_a(zh2, m0, kc * 32, 64);
#pragma unroll
            for (int nt = 0; nt < 4; ++nt) {
                v16h bf = frag_b(wbuf, nt * 16, kc * 32, 64);
                acc[nt] = WMMA_F16(a, bf, acc[nt]);
            }
        }
#pragma unroll
        for (int nt = 0; nt < 4; ++nt)
#pragma unroll
            for (int r8 = 0; r8 < 8; ++r8) {
                int M = m0 + r8 + 8 * mh, O = nt * 16 + Nn;
                outs[M * 64 + O] = acc[nt][r8] + cb2[O];
            }
    }
    __syncthreads();
    for (int idx = tid; idx < 4096; idx += 128) {
        int c = idx >> 6, n = idx & 63;
        y2raw[(size_t)(b * 64 + c) * 16384 + h * 128 + w0 + n] = outs[n * 64 + c];
    }
}

// ---------------- K4: window attention + LAM + proj (one WG per 8x8 window) ----------------
__global__ void k_window_attn(const float* __restrict__ y1raw, const float* __restrict__ y2raw,
                              const float* __restrict__ muY, const float* __restrict__ rsY,
                              const float* __restrict__ g2a, const float* __restrict__ b2a,
                              const float* __restrict__ g2b, const float* __restrict__ b2b,
                              const float* __restrict__ pos1, const float* __restrict__ pos2,
                              const float* __restrict__ qw, const float* __restrict__ qb,
                              const float* __restrict__ kvw, const float* __restrict__ kvb,
                              const float* __restrict__ rpb, const float* __restrict__ gammap,
                              const float* __restrict__ pw, const float* __restrict__ pb,
                              float* __restrict__ xo_t) {
    struct S1 { _Float16 x1h[4096]; _Float16 x2h[4096]; _Float16 wbuf[8192]; };           // 32KB
    struct S2 { _Float16 kh[4096]; _Float16 vT[4096]; _Float16 attnh[4][4096]; };          // 48KB
    struct S3 { _Float16 of[4096]; _Float16 yh[4096]; _Float16 ytb[4096];
                _Float16 xoh[4096]; _Float16 pwh[4096]; _Float16 sE[512]; float eng[256]; };// 42KB
    union U { S1 s1; S2 s2; S3 s3; };
    __shared__ U u;
    __shared__ float qf[4096];  // scaled q, fp32 (attention operand + residual)

    int wg = blockIdx.x;
    int b = wg >> 8;
    int rr = wg & 255;
    int wy = rr >> 4, wx = rr & 15;
    int tid = threadIdx.x;
    int lane = tid & 31;
    int wv = tid >> 5;
    int Nn = lane & 15, mh = lane >> 4;
    float gamma = gammap[0];
    const float scale = 0.25f;  // hd^-0.5, hd=16
    v8f vz = {};

    // ---- Stage 1: build x1w/x2w (gelu(LN2(y)) + pos), f16 token-major ----
    for (int idx = tid; idx < 4096; idx += 128) {
        int c = idx >> 6, n = idx & 63;
        int h = wy * 8 + (n >> 3), w = wx * 8 + (n & 7);
        int hw = h * 128 + w;
        int pc = b * 64 + c;
        size_t gp = (size_t)pc * 16384 + hw;
        float v1 = (y1raw[gp] - muY[pc]) * rsY[pc] * g2a[hw] + b2a[hw];
        u.s1.x1h[n * 64 + c] = (_Float16)(gelu_f(v1) + pos1[n * 64 + c]);
        float v2 = (y2raw[gp] - muY[1024 + pc]) * rsY[1024 + pc] * g2b[hw] + b2b[hw];
        u.s1.x2h[n * 64 + c] = (_Float16)(gelu_f(v2) + pos2[n * 64 + c]);
    }
    for (int idx = tid; idx < 4096; idx += 128) u.s1.wbuf[idx] = (_Float16)qw[idx];
    __syncthreads();

    // ---- q GEMM: qf = scale * (x1w @ qw^T + qb) ----
    {
        int m0 = wv * 16;
        v8f acc[4];
#pragma unroll
        for (int nt = 0; nt < 4; ++nt) acc[nt] = vz;
#pragma unroll
        for (int kc = 0; kc < 2; ++kc) {
            v16h a = frag_a(u.s1.x1h, m0, kc * 32, 64);
#pragma unroll
            for (int nt = 0; nt < 4; ++nt) {
                v16h bf = frag_b(u.s1.wbuf, nt * 16, kc * 32, 64);
                acc[nt] = WMMA_F16(a, bf, acc[nt]);
            }
        }
#pragma unroll
        for (int nt = 0; nt < 4; ++nt)
#pragma unroll
            for (int r8 = 0; r8 < 8; ++r8) {
                int M = m0 + r8 + 8 * mh, O = nt * 16 + Nn;
                qf[M * 64 + O] = scale * (acc[nt][r8] + qb[O]);
            }
    }
    __syncthreads();
    for (int idx = tid; idx < 8192; idx += 128) u.s1.wbuf[idx] = (_Float16)kvw[idx];
    __syncthreads();

    // ---- kv GEMM -> kh (token-major) and vT (channel-major) ----
    {
        int m0 = wv * 16;
        v8f acc[8];
#pragma unroll
        for (int nt = 0; nt < 8; ++nt) acc[nt] = vz;
#pragma unroll
        for (int kc = 0; kc < 2; ++kc) {
            v16h a = frag_a(u.s1.x2h, m0, kc * 32, 64);
#pragma unroll
            for (int nt = 0; nt < 8; ++nt) {
                v16h bf = frag_b(u.s1.wbuf, nt * 16, kc * 32, 64);
                acc[nt] = WMMA_F16(a, bf, acc[nt]);
            }
        }
        __syncthreads();  // all reads of S1 finished: safe to alias as S2
#pragma unroll
        for (int nt = 0; nt < 8; ++nt)
#pragma unroll
            for (int r8 = 0; r8 < 8; ++r8) {
                int M = m0 + r8 + 8 * mh, J = nt * 16 + Nn;
                float v = acc[nt][r8] + kvb[J];
                if (J < 64) u.s2.kh[M * 64 + J] = (_Float16)v;
                else        u.s2.vT[(J - 64) * 64 + M] = (_Float16)v;
            }
    }
    __syncthreads();

    // ---- Attention: wave == head. attn = softmax(q_h @ k_h^T + bias), K=16 padded to 32 ----
    {
        int hh = wv;
#pragma unroll 1
        for (int mt = 0; mt < 4; ++mt) {
            int m0 = mt * 16;
            v8f acc[4];
#pragma unroll
            for (int nt = 0; nt < 4; ++nt) acc[nt] = vz;
            v16h a = frag_a_f32_k16(qf, m0, hh * 16, 64);
#pragma unroll
            for (int nt = 0; nt < 4; ++nt) {
                v16h bf = frag_b_k16(u.s2.kh, nt * 16, hh * 16, 64);
                acc[nt] = WMMA_F16(a, bf, acc[nt]);
            }
            // register-resident softmax: each lane owns 8 rows x 4 cols
#pragma unroll
            for (int r8 = 0; r8 < 8; ++r8) {
                int M = m0 + r8 + 8 * mh;
                int i1 = M >> 3, j1 = M & 7;
                float vals[4];
                float mx = -3.0e38f;
#pragma unroll
                for (int nt = 0; nt < 4; ++nt) {
                    int N = nt * 16 + Nn;
                    int ridx = (i1 - (N >> 3) + 7) * 15 + (j1 - (N & 7) + 7);
                    float vv = acc[nt][r8] + rpb[ridx * 4 + hh];
                    vals[nt] = vv;
                    mx = fmaxf(mx, vv);
                }
#pragma unroll
                for (int d = 8; d >= 1; d >>= 1) mx = fmaxf(mx, __shfl_xor(mx, d, 32));
                float sm = 0.f;
#pragma unroll
                for (int nt = 0; nt < 4; ++nt) { vals[nt] = expf(vals[nt] - mx); sm += vals[nt]; }
#pragma unroll
                for (int d = 8; d >= 1; d >>= 1) sm += __shfl_xor(sm, d, 32);
                float inv = 1.0f / sm;
#pragma unroll
                for (int nt = 0; nt < 4; ++nt)
                    u.s2.attnh[hh][M * 64 + nt * 16 + Nn] = (_Float16)(vals[nt] * inv);
            }
        }
    }
    __syncthreads();

    // ---- o = attn @ v (per head), then rearrange into of / yh / ytb ----
    {
        int hh = wv;
        v8f oacc[4];
#pragma unroll
        for (int mt = 0; mt < 4; ++mt) oacc[mt] = vz;
#pragma unroll
        for (int kc = 0; kc < 2; ++kc) {
            v16h bf = frag_b(u.s2.vT, hh * 16, kc * 32, 64);
#pragma unroll
            for (int mt = 0; mt < 4; ++mt) {
                v16h a = frag_a(u.s2.attnh[hh], mt * 16, kc * 32, 64);
                oacc[mt] = WMMA_F16(a, bf, oacc[mt]);
            }
        }
        __syncthreads();  // all reads of S2 finished: safe to alias as S3
#pragma unroll
        for (int mt = 0; mt < 4; ++mt)
#pragma unroll
            for (int r8 = 0; r8 < 8; ++r8) {
                int M = mt * 16 + r8 + 8 * mh;                 // token n
                _Float16 hv = (_Float16)oacc[mt][r8];          // o[n][16*hh + Nn]
                u.s3.of[M * 64 + hh * 16 + Nn] = hv;
                u.s3.yh[Nn * 256 + hh * 64 + M] = hv;          // y[d][h*64+n]
                u.s3.ytb[(hh * 64 + M) * 16 + Nn] = hv;        // y^T[j][d]
            }
    }
    __syncthreads();

    // ---- LAM energy = y @ y^T (16x16, K=256), wave 0 ----
    if (wv == 0) {
        v8f e = vz;
#pragma unroll
        for (int kc = 0; kc < 8; ++kc) {
            v16h a = frag_a(u.s3.yh, 0, kc * 32, 256);
            v16h bf = frag_b(u.s3.yh, 0, kc * 32, 256);
            e = WMMA_F16(a, bf, e);
        }
#pragma unroll
        for (int r8 = 0; r8 < 8; ++r8) u.s3.eng[(r8 + 8 * mh) * 16 + Nn] = e[r8];
    }
    for (int idx = tid; idx < 512; idx += 128) u.s3.sE[idx] = (_Float16)0.0f;
    __syncthreads();
    if (tid < 16) {
        float mx = -3.0e38f;
        for (int j = 0; j < 16; ++j) mx = fmaxf(mx, u.s3.eng[tid * 16 + j]);
        float ex[16]; float sm = 0.f;
        for (int j = 0; j < 16; ++j) { ex[j] = expf(u.s3.eng[tid * 16 + j] - mx); sm += ex[j]; }
        float inv = 1.0f / sm;
        for (int j = 0; j < 16; ++j) u.s3.sE[tid * 32 + j] = (_Float16)(ex[j] * inv);
    }
    for (int idx = tid; idx < 4096; idx += 128) u.s3.pwh[idx] = (_Float16)pw[idx];
    __syncthreads();

    // ---- lam = softmax(E) @ y ; xo = gamma*lam + y + q + x1w (x1w recomputed) ----
    {
        v16h a = frag_a(u.s3.sE, 0, 0, 32);
#pragma unroll 1
        for (int ntl = 0; ntl < 4; ++ntl) {
            int nt = wv * 4 + ntl;  // 16 N-tiles over 256 cols
            v16h bf = frag_b_k16(u.s3.ytb, nt * 16, 0, 16);
            v8f acc = WMMA_F16(a, bf, vz);
#pragma unroll
            for (int r8 = 0; r8 < 8; ++r8) {
                int i = r8 + 8 * mh;           // head-dim index 0..15
                int J = nt * 16 + Nn;          // 0..255
                int hh = J >> 6, n = J & 63;
                int c = hh * 16 + i;
                float val = gamma * acc[r8] + (float)u.s3.of[n * 64 + c];
                val += qf[n * 64 + c];
                int h = wy * 8 + (n >> 3), w = wx * 8 + (n & 7);
                int hw = h * 128 + w;
                int pc = b * 64 + c;
                float x1v = (y1raw[(size_t)pc * 16384 + hw] - muY[pc]) * rsY[pc] * g2a[hw] + b2a[hw];
                val += gelu_f(x1v) + pos1[n * 64 + c];
                u.s3.xoh[n * 64 + c] = (_Float16)val;
            }
        }
    }
    __syncthreads();

    // ---- proj GEMM -> token-major global output ----
    {
        int m0 = wv * 16;
        v8f acc[4];
#pragma unroll
        for (int nt = 0; nt < 4; ++nt) acc[nt] = vz;
#pragma unroll
        for (int kc = 0; kc < 2; ++kc) {
            v16h a = frag_a(u.s3.xoh, m0, kc * 32, 64);
#pragma unroll
            for (int nt = 0; nt < 4; ++nt) {
                v16h bf = frag_b(u.s3.pwh, nt * 16, kc * 32, 64);
                acc[nt] = WMMA_F16(a, bf, acc[nt]);
            }
        }
#pragma unroll
        for (int nt = 0; nt < 4; ++nt)
#pragma unroll
            for (int r8 = 0; r8 < 8; ++r8) {
                int M = m0 + r8 + 8 * mh;  // token n
                int O = nt * 16 + Nn;
                int h = wy * 8 + (M >> 3), w = wx * 8 + (M & 7);
                xo_t[((size_t)b * 16384 + h * 128 + w) * 64 + O] = acc[nt][r8] + pb[O];
            }
    }
}

// ---------------- K5: token LN + MLP + residual, transpose store to (B,C,H,W) ----------------
__global__ void k_mlp(const float* __restrict__ xo_t,
                      const float* __restrict__ ng, const float* __restrict__ nb,
                      const float* __restrict__ f1w, const float* __restrict__ f1b,
                      const float* __restrict__ f2w, const float* __restrict__ f2b,
                      float* __restrict__ out) {
    __shared__ float ts[4096];
    __shared__ _Float16 tnh[4096];
    __shared__ _Float16 uh[4096];
    __shared__ _Float16 wbuf[4096];
    __shared__ float outs[4096];
    __shared__ float mu_s[64];
    __shared__ float rs_s[64];

    int wg = blockIdx.x;
    int t0 = wg * 64;
    int b = t0 >> 14;
    int hw0 = t0 & 16383;
    int tid = threadIdx.x;
    int lane = tid & 31;
    int wv = tid >> 5;
    int Nn = lane & 15, mh = lane >> 4;
    v8f vz = {};

    for (int idx = tid; idx < 4096; idx += 128) ts[idx] = xo_t[(size_t)t0 * 64 + idx];
    __syncthreads();

    // per-token LN stats (two threads per token)
    {
        int tk = tid >> 1, half = tid & 1;
        float s = 0.f, ss = 0.f;
        for (int c = half * 32; c < half * 32 + 32; ++c) {
            float v = ts[tk * 64 + c];
            s += v; ss += v * v;
        }
        s += __shfl_xor(s, 1, 32);
        ss += __shfl_xor(ss, 1, 32);
        if (half == 0) {
            float m = s * (1.0f / 64.0f);
            float var = ss * (1.0f / 64.0f) - m * m;
            mu_s[tk] = m;
            rs_s[tk] = rsqrtf(var + 1e-5f);
        }
    }
    __syncthreads();
    for (int idx = tid; idx < 4096; idx += 128) {
        int n = idx >> 6, c = idx & 63;
        tnh[idx] = (_Float16)((ts[idx] - mu_s[n]) * rs_s[n] * ng[c] + nb[c]);
    }
    for (int idx = tid; idx < 4096; idx += 128) wbuf[idx] = (_Float16)f1w[idx];
    __syncthreads();

    int m0 = wv * 16;
    // fc1 + GELU
    {
        v8f acc[4];
#pragma unroll
        for (int nt = 0; nt < 4; ++nt) acc[nt] = vz;
#pragma unroll
        for (int kc = 0; kc < 2; ++kc) {
            v16h a = frag_a(tnh, m0, kc * 32, 64);
#pragma unroll
            for (int nt = 0; nt < 4; ++nt) {
                v16h bf = frag_b(wbuf, nt * 16, kc * 32, 64);
                acc[nt] = WMMA_F16(a, bf, acc[nt]);
            }
        }
#pragma unroll
        for (int nt = 0; nt < 4; ++nt)
#pragma unroll
            for (int r8 = 0; r8 < 8; ++r8) {
                int M = m0 + r8 + 8 * mh, O = nt * 16 + Nn;
                uh[M * 64 + O] = (_Float16)gelu_f(acc[nt][r8] + f1b[O]);
            }
    }
    __syncthreads();
    for (int idx = tid; idx < 4096; idx += 128) wbuf[idx] = (_Float16)f2w[idx];
    __syncthreads();
    // fc2 + residual
    {
        v8f acc[4];
#pragma unroll
        for (int nt = 0; nt < 4; ++nt) acc[nt] = vz;
#pragma unroll
        for (int kc = 0; kc < 2; ++kc) {
            v16h a = frag_a(uh, m0, kc * 32, 64);
#pragma unroll
            for (int nt = 0; nt < 4; ++nt) {
                v16h bf = frag_b(wbuf, nt * 16, kc * 32, 64);
                acc[nt] = WMMA_F16(a, bf, acc[nt]);
            }
        }
#pragma unroll
        for (int nt = 0; nt < 4; ++nt)
#pragma unroll
            for (int r8 = 0; r8 < 8; ++r8) {
                int M = m0 + r8 + 8 * mh, O = nt * 16 + Nn;
                outs[M * 64 + O] = ts[M * 64 + O] + acc[nt][r8] + f2b[O];
            }
    }
    __syncthreads();
    // coalesced transposed store to (B,C,H,W)
    int h = hw0 >> 7;
    int w0 = hw0 & 127;
    for (int idx = tid; idx < 4096; idx += 128) {
        int c = idx >> 6, n = idx & 63;
        out[(size_t)(b * 64 + c) * 16384 + h * 128 + w0 + n] = outs[n * 64 + c];
    }
}

// ---------------- host launcher ----------------
extern "C" void kernel_launch(void* const* d_in, const int* in_sizes, int n_in,
                              void* d_out, int out_size, void* d_ws, size_t ws_size,
                              hipStream_t stream) {
    (void)in_sizes; (void)n_in; (void)out_size; (void)ws_size;

    const float* x    = (const float*)d_in[0];
    const float* g1a  = (const float*)d_in[1];
    const float* b1a  = (const float*)d_in[2];
    const float* cw1  = (const float*)d_in[3];
    const float* cb1  = (const float*)d_in[4];
    const float* g2a  = (const float*)d_in[5];
    const float* b2a  = (const float*)d_in[6];
    const float* g1b  = (const float*)d_in[7];
    const float* b1b  = (const float*)d_in[8];
    const float* cw2  = (const float*)d_in[9];
    const float* cb2  = (const float*)d_in[10];
    const float* g2b  = (const float*)d_in[11];
    const float* b2b  = (const float*)d_in[12];
    const float* pos1 = (const float*)d_in[13];
    const float* pos2 = (const float*)d_in[14];
    const float* qw   = (const float*)d_in[15];
    const float* qb   = (const float*)d_in[16];
    const float* kvw  = (const float*)d_in[17];
    const float* kvb  = (const float*)d_in[18];
    const float* rpb  = (const float*)d_in[19];
    const float* gam  = (const float*)d_in[20];
    const float* pw   = (const float*)d_in[21];
    const float* pb   = (const float*)d_in[22];
    const float* ng   = (const float*)d_in[23];
    const float* nb   = (const float*)d_in[24];
    const float* f1w  = (const float*)d_in[25];
    const float* f1b  = (const float*)d_in[26];
    const float* f2w  = (const float*)d_in[27];
    const float* f2b  = (const float*)d_in[28];

    float* ws = (float*)d_ws;
    const size_t PLANE = 16777216;  // 16*64*128*128
    float* y1raw = ws;
    float* y2raw = y1raw + PLANE;
    float* xo_t  = y2raw + PLANE;
    float* mu1   = xo_t + PLANE;
    float* rs1   = mu1 + 1024;
    float* muY   = rs1 + 1024;
    float* rsY   = muY + 2048;

    k_plane_stats<<<1024, 256, 0, stream>>>(x, mu1, rs1, 16384);
    k_lclg_conv<<<4096, 128, 0, stream>>>(x, g1a, b1a, cw1, cb1, g1b, b1b, cw2, cb2,
                                          mu1, rs1, y1raw, y2raw);
    k_plane_stats<<<1024, 256, 0, stream>>>(y1raw, muY, rsY, 16384);
    k_plane_stats<<<1024, 256, 0, stream>>>(y2raw, muY + 1024, rsY + 1024, 16384);
    k_window_attn<<<4096, 128, 0, stream>>>(y1raw, y2raw, muY, rsY, g2a, b2a, g2b, b2b,
                                            pos1, pos2, qw, qb, kvw, kvb, rpb, gam, pw, pb, xo_t);
    k_mlp<<<4096, 128, 0, stream>>>(xo_t, ng, nb, f1w, f1b, f2w, f2b, (float*)d_out);
}
